// DecoderLSTM_79001628443276
// MI455X (gfx1250) — matrix operand order repaired and checked by reading, compiled, not run on hardware
//
#include <hip/hip_runtime.h>

#define B_    4096
#define T_    128
#define V_    30
#define E_    256
#define H_    128
#define G4H   512
#define TSTEPS 127

typedef __attribute__((ext_vector_type(16))) __bf16 v16bf;
typedef __attribute__((ext_vector_type(8)))  float  v8f;

static __device__ __forceinline__ __bf16 f2bf(float f) {
    unsigned u = __builtin_bit_cast(unsigned, f);
    u += 0x7FFFu + ((u >> 16) & 1u);           // round-to-nearest-even
    unsigned short h = (unsigned short)(u >> 16);
    return __builtin_bit_cast(__bf16, h);
}
// fast sigmoid / tanh on the VALU trans pipe: v_exp + v_rcp
static __device__ __forceinline__ float sigm(float x) {
    return __builtin_amdgcn_rcpf(1.0f + __expf(-x));
}
static __device__ __forceinline__ float tanh_f(float x) {
    return 2.0f * __builtin_amdgcn_rcpf(1.0f + __expf(-2.0f * x)) - 1.0f;
}
// swap bits 3<->4 of an index: maps column order so one lane's 16 A-fragment
// elements (K 0..7,16..23 | 8..15,24..31 per ISA layout) become contiguous in LDS
static __device__ __forceinline__ int swz34(int i) {
    return (i & ~0x18) | ((i & 8) << 1) | ((i & 16) >> 1);
}

// ---------------- small setup kernels ----------------

__global__ void zero_acc(float* ws) { ws[0] = 0.0f; ws[1] = 0.0f; }

__global__ void mask_sum_k(const float* __restrict__ m, float* __restrict__ out) {
    float s = 0.0f;
    for (int i = blockIdx.x * blockDim.x + threadIdx.x; i < B_ * T_; i += gridDim.x * blockDim.x)
        s += m[i];
    #pragma unroll
    for (int sh = 16; sh > 0; sh >>= 1) s += __shfl_xor(s, sh);
    if ((threadIdx.x & 31) == 0) atomicAdd(out, s);
}

// P[v][j] = emb[v,:] . W_ih[j,:] + b_ih[j] + b_hh[j]   (30 x 512, eliminates the big input GEMM)
__global__ void precomp_p(const float* __restrict__ emb, const float* __restrict__ Wih,
                          const float* __restrict__ bih, const float* __restrict__ bhh,
                          float* __restrict__ P) {
    int idx = blockIdx.x * 256 + threadIdx.x;
    if (idx >= V_ * G4H) return;
    int v = idx / G4H, j = idx % G4H;
    const float* e = emb + v * E_;
    const float* w = Wih + j * E_;
    float s = 0.0f;
    for (int k = 0; k < E_; ++k) s = fmaf(e[k], w[k], s);
    P[idx] = s + bih[j] + bhh[j];
}

__global__ void conv_whh(const float* __restrict__ Whh, __bf16* __restrict__ out) {
    int idx = blockIdx.x * 256 + threadIdx.x;
    if (idx < G4H * H_) out[idx] = f2bf(Whh[idx]);
}

// W_lin padded to 32 vocab rows (rows 30..31 = 0) in bf16, B-fragment-ready row-major [32][128]
__global__ void conv_wlin(const float* __restrict__ Wlin, __bf16* __restrict__ out) {
    int idx = blockIdx.x * 256 + threadIdx.x;
    if (idx >= 32 * H_) return;
    out[idx] = (idx < V_ * H_) ? f2bf(Wlin[idx]) : f2bf(0.0f);
}

// ---------------- main recurrent kernel ----------------
// 64 workgroups x 256 threads (8 waves). Each WG owns 64 batch rows for all 127 steps.
// Hidden GEMM: wave w owns gate columns [16w,16w+16) -> i/f/g/o fragments + c-state stay
// in registers the whole time. Logits GEMM: wave w owns tile (M=w&3, N=w>>2) of [64,32].
// hLds is stored row-major (no swizzle); contiguous k order makes A fragments single v16bf loads.

#define SM_WHH    0                        // 512*128 bf16  = 131072
#define SM_P      131072                   // 30*512 f32    =  61440
#define SM_WLIN   192512                   // 32*128 bf16   =   8192
#define SM_BLIN   200704                   // 32 f32        =    128
#define SM_H      200832                   // 64*128 bf16   =  16384
#define SM_LOGIT  217216                   // 64*32 f32     =   8192
#define SM_TOK    225408                   // 64 int
#define SM_TGT    225664                   // 64 int
#define SM_RED    225920                   // 8 f32
#define SM_BYTES  226048

__global__ __launch_bounds__(256) void lstm_main(
    const int* __restrict__ inpt, const float* __restrict__ h0, const float* __restrict__ c0,
    const float* __restrict__ maskY, const float* __restrict__ P,
    const __bf16* __restrict__ Whh_bf, const __bf16* __restrict__ Wlin_bf,
    const float* __restrict__ blin, float* __restrict__ ceAcc)
{
    extern __shared__ char smem[];
    __bf16* wLds    = (__bf16*)(smem + SM_WHH);
    float*  pLds    = (float*)(smem + SM_P);
    __bf16* wlinLds = (__bf16*)(smem + SM_WLIN);
    float*  blLds   = (float*)(smem + SM_BLIN);
    __bf16* hLds    = (__bf16*)(smem + SM_H);
    float*  logitLds= (float*)(smem + SM_LOGIT);
    int*    tokLds  = (int*)(smem + SM_TOK);
    int*    tgtLds  = (int*)(smem + SM_TGT);
    float*  redLds  = (float*)(smem + SM_RED);

    const int tid    = threadIdx.x;
    const int wid    = tid >> 5;          // wave 0..7
    const int lane   = tid & 31;
    const int lane16 = lane & 15;
    const int upper  = lane >> 4;
    const int hi8    = upper ? 8 : 0;     // C/D fragment: row offset for lanes 16..31
    const int offA   = upper ? 16 : 0;    // A fragment: K base for lanes 16..31 (k = 16h + s)
    const int kbB    = upper ? 16 : 0;    // B fragment: K base, contiguous k = 16h + s
    const int wgbase = blockIdx.x * 64;
    const int colSw  = 16 * wid + lane16;          // hidden column for h2 stores

    // ---- stage weights / tables / initial h into LDS ----
    for (int i = tid; i < G4H * H_; i += 256) wLds[i] = Whh_bf[i];
    for (int i = tid; i < V_ * G4H; i += 256) pLds[i] = P[i];
    for (int i = tid; i < 32 * H_; i += 256) wlinLds[i] = Wlin_bf[i];
    if (tid < 32) blLds[tid] = (tid < V_) ? blin[tid] : -1e30f;  // padded vocab -> -inf
    for (int i = tid; i < 64 * H_; i += 256) hLds[i] = f2bf(h0[wgbase * H_ + i]);
    __syncthreads();

    // ---- c-state resident in registers, fragment layout ----
    v8f cst[4];
    #pragma unroll
    for (int mt = 0; mt < 4; ++mt)
        #pragma unroll
        for (int v = 0; v < 8; ++v)
            cst[mt][v] = c0[(wgbase + 16 * mt + v + hi8) * H_ + 16 * wid + lane16];

    float ce = 0.0f;
    const int r4 = tid >> 2;     // batch row for softmax phase
    const int q4 = tid & 3;      // vocab chunk of 8
    const int lmt = wid & 3;     // logits tile: M (batch) tile
    const int lnt = wid >> 2;    // logits tile: N (vocab) tile

    for (int t = 0; t < TSTEPS; ++t) {
        if (tid < 64) {
            tokLds[tid] = inpt[(wgbase + tid) * T_ + t];
            tgtLds[tid] = inpt[(wgbase + tid) * T_ + t + 1];
        }
        __syncthreads();                                   // (A)

        // gates init: gather P[token] (replaces x @ W_ih^T + biases)
        v8f acc[4][4];
        #pragma unroll
        for (int mt = 0; mt < 4; ++mt)
            #pragma unroll
            for (int g = 0; g < 4; ++g)
                #pragma unroll
                for (int v = 0; v < 8; ++v) {
                    int row = 16 * mt + v + hi8;
                    acc[mt][g][v] = pLds[tokLds[row] * G4H + g * H_ + 16 * wid + lane16];
                }

        // gates += h @ W_hh^T  via bf16 WMMA (4 K-steps of 32)
        #pragma unroll
        for (int ks = 0; ks < 4; ++ks) {
            v16bf afr[4];
            #pragma unroll
            for (int mt = 0; mt < 4; ++mt)
                afr[mt] = *(const v16bf*)&hLds[(16 * mt + lane16) * H_ + ks * 32 + offA];
            #pragma unroll
            for (int g = 0; g < 4; ++g) {
                int j = g * H_ + 16 * wid + lane16;          // W_hh row (gate output index)
                v16bf bfr = *(const v16bf*)&wLds[j * H_ + ks * 32 + kbB];
                #pragma unroll
                for (int mt = 0; mt < 4; ++mt)
                    acc[mt][g] = __builtin_amdgcn_wmma_f32_16x16x32_bf16(
                        false, afr[mt], false, bfr, (short)0, acc[mt][g], false, false);
            }
        }
        __syncthreads();                                   // (B) done reading h(t-1)

        // LSTM cell update: element-wise, fragments already aligned across gates
        #pragma unroll
        for (int mt = 0; mt < 4; ++mt)
            #pragma unroll
            for (int v = 0; v < 8; ++v) {
                float ig = sigm(acc[mt][0][v]);
                float fg = sigm(acc[mt][1][v]);
                float gg = tanh_f(acc[mt][2][v]);
                float og = sigm(acc[mt][3][v]);
                float c2 = fg * cst[mt][v] + ig * gg;
                cst[mt][v] = c2;
                float h2 = og * tanh_f(c2);
                hLds[(16 * mt + v + hi8) * H_ + colSw] = f2bf(h2);
            }
        __syncthreads();                                   // (C) h(t) visible

        // logits = h2 @ W_lin^T + b_lin  via WMMA: one 16x16 tile per wave
        {
            v8f lacc;
            float bias = blLds[16 * lnt + lane16];
            #pragma unroll
            for (int v = 0; v < 8; ++v) lacc[v] = bias;
            #pragma unroll
            for (int ks = 0; ks < 4; ++ks) {
                v16bf a = *(const v16bf*)&hLds[(16 * lmt + lane16) * H_ + ks * 32 + offA];
                v16bf b = *(const v16bf*)&wlinLds[(16 * lnt + lane16) * H_ + ks * 32 + kbB];
                lacc = __builtin_amdgcn_wmma_f32_16x16x32_bf16(
                    false, a, false, b, (short)0, lacc, false, false);
            }
            #pragma unroll
            for (int v = 0; v < 8; ++v)
                logitLds[(16 * lmt + v + hi8) * 32 + 16 * lnt + lane16] = lacc[v];
        }
        __syncthreads();                                   // (D) logits visible

        // log_softmax + CE: 4 lanes per batch row, 8 vocab entries each (padded to 32)
        {
            const float* lp = &logitLds[r4 * 32 + q4 * 8];
            float l[8];
            float4 p0 = *(const float4*)lp;
            float4 p1 = *(const float4*)(lp + 4);
            l[0] = p0.x; l[1] = p0.y; l[2] = p0.z; l[3] = p0.w;
            l[4] = p1.x; l[5] = p1.y; l[6] = p1.z; l[7] = p1.w;
            float mx = l[0];
            #pragma unroll
            for (int vv = 1; vv < 8; ++vv) mx = fmaxf(mx, l[vv]);
            mx = fmaxf(mx, __shfl_xor(mx, 1));
            mx = fmaxf(mx, __shfl_xor(mx, 2));
            float se = 0.0f;
            #pragma unroll
            for (int vv = 0; vv < 8; ++vv) se += __expf(l[vv] - mx);
            se += __shfl_xor(se, 1);
            se += __shfl_xor(se, 2);
            float lse = mx + __logf(se);
            int y = tgtLds[r4];
            if ((y >> 3) == q4) {
                float m = maskY[(wgbase + r4) * T_ + t];
                ce += m * (lse - l[y & 7]);
            }
        }
        __syncthreads();                                   // protect tokLds/hLds/logitLds
    }

    // reduce CE across the workgroup and accumulate globally
    #pragma unroll
    for (int s = 16; s > 0; s >>= 1) ce += __shfl_xor(ce, s);
    if (lane == 0) redLds[wid] = ce;
    __syncthreads();
    if (tid == 0) {
        float s = 0.0f;
        #pragma unroll
        for (int wv = 0; wv < 8; ++wv) s += redLds[wv];
        atomicAdd(ceAcc, s);
    }
}

__global__ void finalize_k(const float* __restrict__ ws, float* __restrict__ out) {
    out[0] = ws[0] / ws[1];
}

// ---------------- host launcher ----------------

extern "C" void kernel_launch(void* const* d_in, const int* in_sizes, int n_in,
                              void* d_out, int out_size, void* d_ws, size_t ws_size,
                              hipStream_t stream)
{
    const int*   inpt = (const int*)d_in[0];
    const float* h0   = (const float*)d_in[1];
    const float* c0   = (const float*)d_in[2];
    const float* mask = (const float*)d_in[3];
    /* d_in[4] = beta (unused by reference body) */
    const float* emb  = (const float*)d_in[5];
    const float* Wih  = (const float*)d_in[6];
    const float* bih  = (const float*)d_in[7];
    const float* Whh  = (const float*)d_in[8];
    const float* bhh  = (const float*)d_in[9];
    const float* Wlin = (const float*)d_in[10];
    const float* blin = (const float*)d_in[11];

    char*   ws     = (char*)d_ws;
    float*  acc    = (float*)ws;                             // [0]=ce_sum, [1]=mask_sum
    float*  P      = (float*)(ws + 256);                     // 30*512 f32  = 61440 B
    __bf16* whhbf  = (__bf16*)(ws + 256 + 61440);            // 512*128 bf16 = 131072 B
    __bf16* wlinbf = (__bf16*)(ws + 256 + 61440 + 131072);   // 32*128 bf16 = 8192 B

    zero_acc<<<1, 1, 0, stream>>>(acc);
    mask_sum_k<<<512, 256, 0, stream>>>(mask, acc + 1);
    precomp_p<<<(V_ * G4H + 255) / 256, 256, 0, stream>>>(emb, Wih, bih, bhh, P);
    conv_whh<<<(G4H * H_ + 255) / 256, 256, 0, stream>>>(Whh, whhbf);
    conv_wlin<<<(32 * H_ + 255) / 256, 256, 0, stream>>>(Wlin, wlinbf);
    lstm_main<<<64, 256, SM_BYTES, stream>>>(inpt, h0, c0, mask, P, whhbf, wlinbf, blin, acc);
    finalize_k<<<1, 1, 0, stream>>>(acc, (float*)d_out);
}
